// HierarchicalAttention_26577257627870
// MI455X (gfx1250) — compile-verified, matrix operand order checked by reading, not run on hardware
//
#include <hip/hip_runtime.h>
#include <hip/hip_bf16.h>

// ---------------------------------------------------------------------------
// Windowed multi-head attention for MI455X (gfx1250), wave32 + WMMA.
//   B=4, N=8192, C=1024, H=16, D=64, WINDOW=64
// Pipeline:
//   [0] f32 -> f16 converts (x, w_qkv, w_proj)
//   [1] qkv  = x @ w_qkv^T + b_qkv      (WMMA f32_16x16x32_f16, f16 out)
//   [2] per-window attention (QK^T -> softmax -> PV), WMMA + LDS transpose
//   [3] out  = attn @ w_proj^T + b_proj (WMMA, f32 out)
// ---------------------------------------------------------------------------

typedef _Float16 h16;
typedef __attribute__((ext_vector_type(16))) _Float16 v16h;
typedef __attribute__((ext_vector_type(8)))  _Float16 v8h;
typedef __attribute__((ext_vector_type(8)))  float    v8f;

#define DIMC    1024
#define NHEADS  16
#define HEADD   64
#define WIN     64
#define QKVCOLS 3072

__device__ __forceinline__ v16h make_frag(v8h lo, v8h hi) {
    union { v16h v; v8h h[2]; } u;
    u.h[0] = lo; u.h[1] = hi;
    return u.v;
}

// Load one 16x32 f16 WMMA operand fragment for this lane from a row-major
// matrix whose K dimension is contiguous. Per the CDNA5 ISA 16-bit A/B
// layout: lanes 0-15 hold K = {k0+0..7, k0+16..23}, lanes 16-31 hold
// K = {k0+8..15, k0+24..31}; two 16-byte loads per fragment.
__device__ __forceinline__ v16h load_frag(const h16* row, int k0, int lh) {
    v8h lo = *reinterpret_cast<const v8h*>(row + k0 + lh * 8);
    v8h hi = *reinterpret_cast<const v8h*>(row + k0 + 16 + lh * 8);
    return make_frag(lo, hi);
}

__device__ __forceinline__ v8f wmma_f16(v16h a, v16h b, v8f c) {
    return __builtin_amdgcn_wmma_f32_16x16x32_f16(
        /*neg_a=*/false, a, /*neg_b=*/false, b,
        /*c_mod=*/(short)0, c, /*reuse_a=*/false, /*reuse_b=*/false);
}

// ------------------------------ converts -----------------------------------
__global__ void f32_to_f16_kernel(const float* __restrict__ src,
                                  h16* __restrict__ dst, size_t n) {
    size_t i = (size_t)blockIdx.x * blockDim.x + threadIdx.x;
    size_t stride = (size_t)gridDim.x * blockDim.x;
    for (; i < n; i += stride) dst[i] = (h16)src[i];
}

// ------------------------------ WMMA GEMM ----------------------------------
// C[M,N] = A[M,K] * Bw[N,K]^T + bias[N]
// One wave -> 64x64 output tile (4x4 WMMA accumulators), 4 waves/block ->
// 128x128 block tile. M,N multiples of 128; K multiple of 32.
template <bool OUT_F16>
__global__ __launch_bounds__(128)
void gemm_wmma_kernel(const h16* __restrict__ A, const h16* __restrict__ Bw,
                      const float* __restrict__ bias, void* __restrict__ Cv,
                      int M, int N, int K) {
    const int lane = threadIdx.x & 31;
    const int wave = threadIdx.x >> 5;
    const int ln = lane & 15;
    const int lh = lane >> 4;

    const int mBase = blockIdx.x * 128 + (wave >> 1) * 64;
    const int nBase = blockIdx.y * 128 + (wave & 1) * 64;

    const h16* aRow[4];
    const h16* bRow[4];
#pragma unroll
    for (int t = 0; t < 4; ++t) {
        aRow[t] = A  + (size_t)(mBase + t * 16 + ln) * K;
        bRow[t] = Bw + (size_t)(nBase + t * 16 + ln) * K;
    }

    v8f acc[4][4] = {};

    for (int k0 = 0; k0 < K; k0 += 32) {
        v16h af[4], bf[4];
#pragma unroll
        for (int t = 0; t < 4; ++t) af[t] = load_frag(aRow[t], k0, lh);
#pragma unroll
        for (int t = 0; t < 4; ++t) bf[t] = load_frag(bRow[t], k0, lh);
#pragma unroll
        for (int mt = 0; mt < 4; ++mt)
#pragma unroll
            for (int nt = 0; nt < 4; ++nt)
                acc[mt][nt] = wmma_f16(af[mt], bf[nt], acc[mt][nt]);
    }

    // Epilogue. C/D layout: lane half lh, column n = ln (+tile), VGPR r ->
    // row m = 8*lh + r (+tile).
#pragma unroll
    for (int nt = 0; nt < 4; ++nt) {
        const int n = nBase + nt * 16 + ln;
        const float bv = bias[n];
#pragma unroll
        for (int mt = 0; mt < 4; ++mt) {
            const int mb = mBase + mt * 16 + 8 * lh;
#pragma unroll
            for (int r = 0; r < 8; ++r) {
                const float v = acc[mt][nt][r] + bv;
                const size_t idx = (size_t)(mb + r) * N + n;
                if (OUT_F16) ((h16*)Cv)[idx] = (h16)v;
                else         ((float*)Cv)[idx] = v;
            }
        }
    }
}

// --------------------------- window attention ------------------------------
// One wave per (batch, head, window). qkv: [B*N, 3072] f16 row-major,
// columns = which*1024 + head*64 + d. out: [B*N, 1024] f16.
__global__ __launch_bounds__(32)
void win_attn_kernel(const h16* __restrict__ qkv, h16* __restrict__ outh) {
    const int idx  = blockIdx.x;
    const int win  = idx & 127;          // N/WIN = 128 windows per (b,h)
    const int head = (idx >> 7) & 15;
    const int b    = idx >> 11;

    const size_t token0 = (size_t)b * 8192 + (size_t)win * WIN;
    const h16* qBase = qkv + token0 * QKVCOLS + head * HEADD;
    const h16* kBase = qBase + DIMC;
    const h16* vBase = qBase + 2 * DIMC;

    const int lane = threadIdx.x;
    const int ln = lane & 15;
    const int lh = lane >> 4;

    __shared__ h16 Pl[64 * 72];   // softmax probs, padded stride
    __shared__ h16 Vt[64 * 72];   // V transposed: Vt[d][j] = V[j][d]

    // Stage V transposed into LDS: each lane pulls 8 contiguous halves of a
    // row and scatters them down a column of Vt.
    for (int jo = 0; jo < 16; ++jo) {
        const int j  = jo * 4 + (lane >> 3);
        const int d0 = (lane & 7) * 8;
        v8h vv = *reinterpret_cast<const v8h*>(vBase + (size_t)j * QKVCOLS + d0);
#pragma unroll
        for (int t = 0; t < 8; ++t) Vt[(d0 + t) * 72 + j] = vv[t];
    }

    // ---- S = Q K^T (64x64x64, two 32-K steps, 32 WMMAs) ----
    const h16* qRow[4];
    const h16* kRow[4];
#pragma unroll
    for (int t = 0; t < 4; ++t) {
        qRow[t] = qBase + (size_t)(t * 16 + ln) * QKVCOLS;
        kRow[t] = kBase + (size_t)(t * 16 + ln) * QKVCOLS;
    }

    v8f acc[4][4] = {};
#pragma unroll
    for (int ks = 0; ks < 2; ++ks) {
        const int k0 = ks * 32;
        v16h af[4], bf[4];
#pragma unroll
        for (int t = 0; t < 4; ++t) af[t] = load_frag(qRow[t], k0, lh);
#pragma unroll
        for (int t = 0; t < 4; ++t) bf[t] = load_frag(kRow[t], k0, lh);
#pragma unroll
        for (int mt = 0; mt < 4; ++mt)
#pragma unroll
            for (int nt = 0; nt < 4; ++nt)
                acc[mt][nt] = wmma_f16(af[mt], bf[nt], acc[mt][nt]);
    }

    // ---- softmax over rows (scale = 1/sqrt(64) = 0.125) ----
    // Row m of tile mt sits in VGPR r across the 16 lanes of one half, so the
    // reduction is: combine the 4 column-tiles locally, then shfl_xor 8/4/2/1.
    const float scale = 0.125f;
#pragma unroll
    for (int mt = 0; mt < 4; ++mt)
#pragma unroll
        for (int nt = 0; nt < 4; ++nt)
#pragma unroll
            for (int r = 0; r < 8; ++r) acc[mt][nt][r] *= scale;

#pragma unroll
    for (int mt = 0; mt < 4; ++mt) {
        float rmax[8], rsum[8];
#pragma unroll
        for (int r = 0; r < 8; ++r) {
            float m0 = -1e30f;
#pragma unroll
            for (int nt = 0; nt < 4; ++nt) m0 = fmaxf(m0, acc[mt][nt][r]);
            m0 = fmaxf(m0, __shfl_xor(m0, 8, 32));
            m0 = fmaxf(m0, __shfl_xor(m0, 4, 32));
            m0 = fmaxf(m0, __shfl_xor(m0, 2, 32));
            m0 = fmaxf(m0, __shfl_xor(m0, 1, 32));
            rmax[r] = m0;
        }
#pragma unroll
        for (int r = 0; r < 8; ++r) {
            float s = 0.f;
#pragma unroll
            for (int nt = 0; nt < 4; ++nt) {
                const float e = __expf(acc[mt][nt][r] - rmax[r]);
                acc[mt][nt][r] = e;
                s += e;
            }
            s += __shfl_xor(s, 8, 32);
            s += __shfl_xor(s, 4, 32);
            s += __shfl_xor(s, 2, 32);
            s += __shfl_xor(s, 1, 32);
            rsum[r] = s;
        }
#pragma unroll
        for (int r = 0; r < 8; ++r) {
            const float inv = 1.0f / rsum[r];
#pragma unroll
            for (int nt = 0; nt < 4; ++nt) {
                const float p = acc[mt][nt][r] * inv;
                Pl[(mt * 16 + 8 * lh + r) * 72 + nt * 16 + ln] = (h16)p;
            }
        }
    }

    __syncthreads();   // LDS writes (Pl, Vt) -> LDS reads below

    // ---- O = P V (A = P rows from LDS, B = Vt rows from LDS) ----
    v8f acc2[4][4] = {};
#pragma unroll
    for (int ks = 0; ks < 2; ++ks) {
        const int k0 = ks * 32;
        v16h af[4], bf[4];
#pragma unroll
        for (int t = 0; t < 4; ++t)
            af[t] = load_frag(&Pl[(t * 16 + ln) * 72], k0, lh);
#pragma unroll
        for (int t = 0; t < 4; ++t)
            bf[t] = load_frag(&Vt[(t * 16 + ln) * 72], k0, lh);
#pragma unroll
        for (int mt = 0; mt < 4; ++mt)
#pragma unroll
            for (int nt = 0; nt < 4; ++nt)
                acc2[mt][nt] = wmma_f16(af[mt], bf[nt], acc2[mt][nt]);
    }

    // Store f16 result into [token, head*64 + d] layout for the proj GEMM.
    h16* oBase = outh + token0 * DIMC + head * HEADD;
#pragma unroll
    for (int nt = 0; nt < 4; ++nt) {
        const int d = nt * 16 + ln;
#pragma unroll
        for (int mt = 0; mt < 4; ++mt) {
            const int ib = mt * 16 + 8 * lh;
#pragma unroll
            for (int r = 0; r < 8; ++r)
                oBase[(size_t)(ib + r) * DIMC + d] = (h16)acc2[mt][nt][r];
        }
    }
}

// ------------------------------- launcher ----------------------------------
extern "C" void kernel_launch(void* const* d_in, const int* in_sizes, int n_in,
                              void* d_out, int out_size, void* d_ws, size_t ws_size,
                              hipStream_t stream) {
    const float* x      = (const float*)d_in[0];
    const float* w_qkv  = (const float*)d_in[1];
    const float* b_qkv  = (const float*)d_in[2];
    const float* w_proj = (const float*)d_in[3];
    const float* b_proj = (const float*)d_in[4];

    const size_t M = (size_t)in_sizes[0] / DIMC;   // B*N = 32768 tokens

    // Workspace carve-up (f16 staging buffers).
    char* ws = (char*)d_ws;
    size_t off = 0;
    h16* xh     = (h16*)(ws + off); off += M * DIMC * sizeof(h16);
    h16* wqkvh  = (h16*)(ws + off); off += (size_t)QKVCOLS * DIMC * sizeof(h16);
    h16* wprojh = (h16*)(ws + off); off += (size_t)DIMC * DIMC * sizeof(h16);
    h16* qkvh   = (h16*)(ws + off); off += M * QKVCOLS * sizeof(h16);
    h16* attnh  = (h16*)(ws + off); off += M * DIMC * sizeof(h16);
    (void)ws_size; (void)n_in; (void)out_size;

    // [0] converts
    f32_to_f16_kernel<<<4096, 256, 0, stream>>>(x, xh, M * DIMC);
    f32_to_f16_kernel<<<1024, 256, 0, stream>>>(w_qkv, wqkvh,
                                                (size_t)QKVCOLS * DIMC);
    f32_to_f16_kernel<<<512, 256, 0, stream>>>(w_proj, wprojh,
                                               (size_t)DIMC * DIMC);

    // [1] qkv = x @ w_qkv^T + b_qkv   (f16 out)
    gemm_wmma_kernel<true><<<dim3((unsigned)(M / 128), QKVCOLS / 128),
                             128, 0, stream>>>(
        xh, wqkvh, b_qkv, qkvh, (int)M, QKVCOLS, DIMC);

    // [2] windowed attention: one wave per (b, head, window)
    const unsigned nWin = (unsigned)(M / WIN) * NHEADS;   // 512 * 16 = 8192
    win_attn_kernel<<<nWin, 32, 0, stream>>>(qkvh, attnh);

    // [3] out = attn @ w_proj^T + b_proj   (f32 out)
    gemm_wmma_kernel<false><<<dim3((unsigned)(M / 128), DIMC / 128),
                              128, 0, stream>>>(
        attnh, wprojh, b_proj, (float*)d_out, (int)M, DIMC, DIMC);
}
